// STACAME_Multi_77644418777399
// MI455X (gfx1250) — matrix operand order repaired and checked by reading, not compiled, not run on hardware
//
#include <hip/hip_runtime.h>
#include <hip/hip_bf16.h>
#include <math.h>

// ---------------------------------------------------------------------------
// CDNA5 (gfx1250) bf16 WMMA GEMM (128x128 block tile, double-buffered LDS)
// + GAT scatter pipeline.
// ---------------------------------------------------------------------------

typedef __attribute__((ext_vector_type(16))) __bf16 v16bf;
typedef __attribute__((ext_vector_type(8)))  float  v8f;

union FragBF {
  uint4 u[2];   // 2 x ds_load_b128
  v16bf v;      // WMMA operand (16 bf16 = 8 VGPRs)
};

__device__ __forceinline__ unsigned short f2bf(float f) {
  unsigned int u = __float_as_uint(f);
  u += 0x7FFFu + ((u >> 16) & 1u);      // round-to-nearest-even
  return (unsigned short)(u >> 16);
}

__device__ __forceinline__ unsigned int pk2bf(float a, float b) {
  return (unsigned int)f2bf(a) | ((unsigned int)f2bf(b) << 16);
}

#define BM 128
#define BN 128
#define BK 32
#define LDSA (BK + 8)   // 40 ushorts = 80B row stride (16B aligned, skewed banks)

// Stage one (A,B) K-chunk into LDS buffers (bf16). Uniform scalar branches only.
__device__ __forceinline__ void stage_tiles(
    unsigned short (*As)[LDSA], unsigned short (*Bs)[LDSA],
    const float* __restrict__ A, const float* __restrict__ B,
    int M, int N, int K, int lda, int ldb, int transB,
    int m0, int n0, int k0, int tid, bool avec, bool bvec) {
  const bool fullK = (k0 + BK <= K);

  // ---------------- A tile: BM x BK ----------------
  if (avec && fullK) {
    // 4096 floats = 4 float4 per thread; b128 loads, b64 LDS stores
#pragma unroll
    for (int i = 0; i < 4; ++i) {
      int idx = i * 256 + tid;          // float4 index
      int r = idx >> 3;                 // 8 float4 per 32-wide row
      int c4 = (idx & 7) << 2;
      const float* src = &A[(size_t)(m0 + r) * lda + k0 + c4];
      if (i == 0) __builtin_prefetch(src + 2 * BK, 0, 1);   // global_prefetch_b8
      float4 v = *reinterpret_cast<const float4*>(src);
      uint2 pk = make_uint2(pk2bf(v.x, v.y), pk2bf(v.z, v.w));
      *reinterpret_cast<uint2*>(&As[r][c4]) = pk;
    }
  } else {
    // clamped-address loads + select: no divergent exec-mask juggling
#pragma unroll
    for (int i = 0; i < 16; ++i) {
      int idx = i * 256 + tid;
      int r = idx >> 5, c = idx & 31;
      int gm = m0 + r, gk = k0 + c;
      int gmc = gm < M ? gm : M - 1;
      int gkc = gk < K ? gk : K - 1;
      float v = A[(size_t)gmc * lda + gkc];
      v = (gm < M && gk < K) ? v : 0.0f;
      As[r][c] = f2bf(v);
    }
  }

  // ---------------- B tile -> Bt[n][k]: BN x BK ----------------
  if (bvec && fullK) {
    if (transB) {                       // read along k (contiguous), no transpose
#pragma unroll
      for (int i = 0; i < 4; ++i) {
        int idx = i * 256 + tid;
        int n = idx >> 3, c4 = (idx & 7) << 2;
        float4 v = *reinterpret_cast<const float4*>(
            &B[(size_t)(n0 + n) * ldb + k0 + c4]);
        uint2 pk = make_uint2(pk2bf(v.x, v.y), pk2bf(v.z, v.w));
        *reinterpret_cast<uint2*>(&Bs[n][c4]) = pk;
      }
    } else {                            // read along n (contiguous), scatter-transpose
#pragma unroll
      for (int i = 0; i < 4; ++i) {
        int idx = i * 256 + tid;
        int k = idx >> 5, n4 = (idx & 31) << 2;
        float4 v = *reinterpret_cast<const float4*>(
            &B[(size_t)(k0 + k) * ldb + n0 + n4]);
        Bs[n4 + 0][k] = f2bf(v.x);
        Bs[n4 + 1][k] = f2bf(v.y);
        Bs[n4 + 2][k] = f2bf(v.z);
        Bs[n4 + 3][k] = f2bf(v.w);
      }
    }
  } else {
#pragma unroll
    for (int i = 0; i < 16; ++i) {
      int idx = i * 256 + tid;
      int n, c;
      if (transB) { n = idx >> 5; c = idx & 31; }
      else        { c = idx >> 7; n = idx & 127; }
      int gn = n0 + n, gk = k0 + c;
      int gnc = gn < N ? gn : N - 1;
      int gkc = gk < K ? gk : K - 1;
      size_t off = transB ? ((size_t)gnc * ldb + gkc) : ((size_t)gkc * ldb + gnc);
      float v = B[off];
      v = (gn < N && gk < K) ? v : 0.0f;
      Bs[n][c] = f2bf(v);
    }
  }
}

// C[M,N] = act(A[M,K] @ B + bias), B is [K,N] (transB=0) or row-major [N,K]^T
// (transB=1). bf16 LDS staging, f32 accumulate, v_wmma_f32_16x16x32_bf16.
// 256 threads = 8 waves; wave (wq,wc) owns a 32x64 sub-tile: 2 A-frags x
// 4 B-frags -> 8 WMMA per K-chunk against 12 ds_load_b128.
__global__ __launch_bounds__(256) void wmma_gemm_kernel(
    const float* __restrict__ A, const float* __restrict__ B,
    const float* __restrict__ bias, float* __restrict__ C,
    int M, int N, int K, int lda, int ldb, int ldc, int transB, int act) {
  __shared__ __attribute__((aligned(16))) unsigned short As[2][BM][LDSA];
  __shared__ __attribute__((aligned(16))) unsigned short Bs[2][BN][LDSA];

  const int tid  = threadIdx.x;
  const int lane = tid & 31;
  const int wave = tid >> 5;
  const int wq   = wave & 3;            // row group: rows [32*wq, 32*wq+32)
  const int wc   = wave >> 2;           // col group: cols [64*wc, 64*wc+64)
  const int m0 = blockIdx.x * BM;
  const int n0 = blockIdx.y * BN;

  const bool fullM = (m0 + BM <= M);
  const bool fullN = (n0 + BN <= N);
  const bool avec = fullM && ((lda & 3) == 0) && ((((size_t)A) & 15) == 0);
  const bool bvec = fullN && ((ldb & 3) == 0) && ((((size_t)B) & 15) == 0);

  v8f acc[2][4];
#pragma unroll
  for (int i = 0; i < 2; ++i)
#pragma unroll
    for (int t = 0; t < 4; ++t)
#pragma unroll
      for (int j = 0; j < 8; ++j) acc[i][t][j] = 0.0f;

  // prologue: stage chunk 0 into buffer 0
  stage_tiles(As[0], Bs[0], A, B, M, N, K, lda, ldb, transB,
              m0, n0, 0, tid, avec, bvec);

  const int arow = (wq << 5) | (lane & 15);
  const int ka = (lane & 16) >> 1;      // 0 or 8
  const int kb = lane & 16;

  int p = 0;
  for (int k0 = 0; k0 < K; k0 += BK) {
    __syncthreads();                    // buffer p is ready for everyone

    // issue next chunk's global loads before consuming current chunk
    if (k0 + BK < K)
      stage_tiles(As[p ^ 1], Bs[p ^ 1], A, B, M, N, K, lda, ldb, transB,
                  m0, n0, k0 + BK, tid, avec, bvec);

    // A fragments (ISA 16-bit 16x32 layout): lanes<16 -> K{0..7,16..23},
    // lanes>=16 -> K{8..15,24..31}
    FragBF a0, a1;
    a0.u[0] = *reinterpret_cast<const uint4*>(&As[p][arow][ka]);
    a0.u[1] = *reinterpret_cast<const uint4*>(&As[p][arow][16 + ka]);
    a1.u[0] = *reinterpret_cast<const uint4*>(&As[p][arow + 16][ka]);
    a1.u[1] = *reinterpret_cast<const uint4*>(&As[p][arow + 16][16 + ka]);

#pragma unroll
    for (int nt = 0; nt < 4; ++nt) {
      const int brow = (wc << 6) | (nt << 4) | (lane & 15);
      FragBF bfr;
      bfr.u[0] = *reinterpret_cast<const uint4*>(&Bs[p][brow][ka]);
      bfr.u[1] = *reinterpret_cast<const uint4*>(&Bs[p][brow][16 + ka]);
      acc[0][nt] = __builtin_amdgcn_wmma_f32_16x16x32_bf16(
          false, a0.v, false, bfr.v, (short)0, acc[0][nt], false, false);
      acc[1][nt] = __builtin_amdgcn_wmma_f32_16x16x32_bf16(
          false, a1.v, false, bfr.v, (short)0, acc[1][nt], false, false);
    }
    p ^= 1;
  }

  // ---- epilogue: C/D layout M = r + 8*(lane>=16), N = lane&15 ----
  const int coll = lane & 15;
  const int moff = (lane & 16) >> 1;    // 0 or 8
  if (fullM && fullN) {
#pragma unroll
    for (int nt = 0; nt < 4; ++nt) {
      int col = n0 + (wc << 6) + (nt << 4) + coll;
      float bv = bias ? bias[col] : 0.0f;
#pragma unroll
      for (int i = 0; i < 2; ++i) {
#pragma unroll
        for (int r = 0; r < 8; ++r) {
          int row = m0 + (wq << 5) + (i << 4) + moff + r;
          float v = acc[i][nt][r] + bv;
          if (act == 1) v = (v > 0.0f) ? v : 0.01f * v;
          C[(size_t)row * ldc + col] = v;
        }
      }
    }
  } else {
#pragma unroll
    for (int nt = 0; nt < 4; ++nt) {
      int col = n0 + (wc << 6) + (nt << 4) + coll;
      if (col >= N) continue;
      float bv = bias ? bias[col] : 0.0f;
#pragma unroll
      for (int i = 0; i < 2; ++i) {
#pragma unroll
        for (int r = 0; r < 8; ++r) {
          int row = m0 + (wq << 5) + (i << 4) + moff + r;
          if (row >= M) continue;
          float v = acc[i][nt][r] + bv;
          if (act == 1) v = (v > 0.0f) ? v : 0.01f * v;
          C[(size_t)row * ldc + col] = v;
        }
      }
    }
  }
}

// ---------------------------------------------------------------------------
// Helpers
// ---------------------------------------------------------------------------

__global__ void fill_f32(float* __restrict__ p, float v, long long n) {
  long long i = (long long)blockIdx.x * blockDim.x + threadIdx.x;
  long long stride = (long long)gridDim.x * blockDim.x;
  for (; i < n; i += stride) p[i] = v;
}

__global__ void elu_inplace(float* __restrict__ p, long long n) {
  long long i = (long long)blockIdx.x * blockDim.x + threadIdx.x;
  long long stride = (long long)gridDim.x * blockDim.x;
  for (; i < n; i += stride) {
    float v = p[i];
    p[i] = (v > 0.0f) ? v : (__expf(v) - 1.0f);
  }
}

// a_src[i] = xs[i,:]·att_src ; a_dst[i] = xs[i,:]·att_dst  (one wave32 per row)
__global__ __launch_bounds__(256) void att_kernel(
    const float* __restrict__ xs, const float* __restrict__ asv,
    const float* __restrict__ adv, float* __restrict__ a_src,
    float* __restrict__ a_dst, int n, int D) {
  int row = (int)((blockIdx.x * (long long)blockDim.x + threadIdx.x) >> 5);
  int lane = threadIdx.x & 31;
  if (row >= n) return;
  const float4* xr = reinterpret_cast<const float4*>(xs + (size_t)row * D);
  const float4* s4 = reinterpret_cast<const float4*>(asv);
  const float4* d4 = reinterpret_cast<const float4*>(adv);
  float s1 = 0.0f, s2 = 0.0f;
  for (int j = lane; j < (D >> 2); j += 32) {
    float4 v = xr[j], a = s4[j], b = d4[j];
    s1 += v.x * a.x + v.y * a.y + v.z * a.z + v.w * a.w;
    s2 += v.x * b.x + v.y * b.y + v.z * b.z + v.w * b.w;
  }
#pragma unroll
  for (int off = 16; off > 0; off >>= 1) {
    s1 += __shfl_xor(s1, off, 32);
    s2 += __shfl_xor(s2, off, 32);
  }
  if (lane == 0) { a_src[row] = s1; a_dst[row] = s2; }
}

// e = sigmoid(a_src[src]+a_dst[dst]); per-dst running max (positive floats ->
// int-compare atomicMax is order-preserving; m initialized to +0.0).
__global__ void edge_logits(const int* __restrict__ ei, const float* __restrict__ asrc,
                            const float* __restrict__ adst, float* __restrict__ e,
                            float* __restrict__ m, int E) {
  int id = blockIdx.x * blockDim.x + threadIdx.x;
  if (id >= E) return;
  int s = ei[id], d = ei[E + id];
  float x = asrc[s] + adst[d];
  float sg = 1.0f / (1.0f + __expf(-x));
  e[id] = sg;
  atomicMax(reinterpret_cast<int*>(m) + d, __float_as_int(sg));
}

__global__ void edge_expsum(const int* __restrict__ ei, const float* __restrict__ m,
                            float* __restrict__ e, float* __restrict__ denom, int E) {
  int id = blockIdx.x * blockDim.x + threadIdx.x;
  if (id >= E) return;
  int d = ei[E + id];
  float ex = __expf(e[id] - m[d]);
  e[id] = ex;
  atomicAdd(denom + d, ex);
}

__global__ void edge_norm(const int* __restrict__ ei, const float* __restrict__ denom,
                          float* __restrict__ e, int E) {
  int id = blockIdx.x * blockDim.x + threadIdx.x;
  if (id >= E) return;
  int d = ei[E + id];
  e[id] = e[id] / (denom[d] + 1e-16f);
}

// out[dst,:] += w[e] * xs[src,:]  (one wave32 per edge; float4 gathers,
// native global_atomic_add_f32 scatter)
__global__ __launch_bounds__(256) void edge_aggregate(
    const int* __restrict__ ei, const float* __restrict__ w,
    const float* __restrict__ xs, float* __restrict__ out, int E, int D) {
  int gw = (int)((blockIdx.x * (long long)blockDim.x + threadIdx.x) >> 5);
  int lane = threadIdx.x & 31;
  if (gw >= E) return;
  int s = ei[gw], d = ei[E + gw];
  float ww = w[gw];
  const float4* xr = reinterpret_cast<const float4*>(xs + (size_t)s * D);
  float* orow = out + (size_t)d * D;
  for (int j = lane; j < (D >> 2); j += 32) {
    float4 v = xr[j];
    int c = j << 2;
    atomicAdd(&orow[c + 0], v.x * ww);
    atomicAdd(&orow[c + 1], v.y * ww);
    atomicAdd(&orow[c + 2], v.z * ww);
    atomicAdd(&orow[c + 3], v.w * ww);
  }
}

// ---------------------------------------------------------------------------
// Launch
// ---------------------------------------------------------------------------

extern "C" void kernel_launch(void* const* d_in, const int* in_sizes, int n_in,
                              void* d_out, int out_size, void* d_ws, size_t ws_size,
                              hipStream_t stream) {
  const float* x1    = (const float*)d_in[0];
  const float* x2    = (const float*)d_in[1];
  const float* w1    = (const float*)d_in[2];
  const float* b1    = (const float*)d_in[3];
  const float* w2    = (const float*)d_in[4];
  const float* b2    = (const float*)d_in[5];
  const float* c1lin = (const float*)d_in[6];
  const float* attS  = (const float*)d_in[7];
  const float* attD  = (const float*)d_in[8];
  const float* c2lin = (const float*)d_in[9];
  const int*   ei    = (const int*)d_in[10];

  const int F_RAW = 2000, IN_DIM = 512, HID = 512, OUT = 30;
  const int n1 = in_sizes[0] / F_RAW;
  const int n2 = in_sizes[1] / F_RAW;
  const int n  = n1 + n2;
  const int E  = in_sizes[10] / 2;

  // d_out regions: h2 [n,30] | h4 [n,512] | features [n,512]
  float* out   = (float*)d_out;
  float* h2    = out;
  float* h4    = out + (size_t)n * OUT;
  float* feats = h4 + (size_t)n * IN_DIM;

  // workspace
  float* ws   = (float*)d_ws;
  float* xs   = ws;                            // [n,512] xs1 / xs3
  float* h    = xs + (size_t)n * HID;          // [n,512] aggregation buffer
  float* asrc = h + (size_t)n * HID;           // [n]
  float* adst = asrc + n;                      // [n]
  float* mbuf = adst + n;                      // [n]
  float* dbuf = mbuf + n;                      // [n]
  float* ew   = dbuf + n;                      // [E] edge weights

  dim3 blk(256);
  auto gx = [](int M) { return (unsigned)((M + BM - 1) / BM); };
  auto gy = [](int N) { return (unsigned)((N + BN - 1) / BN); };
  const unsigned egrid = (unsigned)((E + 255) / 256);
  const unsigned ewave = (unsigned)(((long long)E * 32 + 255) / 256);

  // features = leaky_relu(x @ w + b)   (two species)
  wmma_gemm_kernel<<<dim3(gx(n1), gy(IN_DIM)), blk, 0, stream>>>(
      x1, w1, b1, feats, n1, IN_DIM, F_RAW, F_RAW, IN_DIM, IN_DIM, 0, 1);
  wmma_gemm_kernel<<<dim3(gx(n2), gy(IN_DIM)), blk, 0, stream>>>(
      x2, w2, b2, feats + (size_t)n1 * IN_DIM, n2, IN_DIM, F_RAW, F_RAW, IN_DIM, IN_DIM, 0, 1);

  // xs1 = features @ conv1_lin
  wmma_gemm_kernel<<<dim3(gx(n), gy(HID)), blk, 0, stream>>>(
      feats, c1lin, nullptr, xs, n, HID, IN_DIM, IN_DIM, HID, HID, 0, 0);

  // attention scalars (tied for both GAT layers)
  att_kernel<<<(unsigned)(((long long)n * 32 + 255) / 256), blk, 0, stream>>>(
      xs, attS, attD, asrc, adst, n, HID);

  // softmax over incoming edges — computed ONCE, reused by conv3
  fill_f32<<<2048, blk, 0, stream>>>(mbuf, 0.0f, n);
  fill_f32<<<2048, blk, 0, stream>>>(dbuf, 0.0f, n);
  edge_logits<<<egrid, blk, 0, stream>>>(ei, asrc, adst, ew, mbuf, E);
  edge_expsum<<<egrid, blk, 0, stream>>>(ei, mbuf, ew, dbuf, E);
  edge_norm<<<egrid, blk, 0, stream>>>(ei, dbuf, ew, E);

  // h1 = elu(aggregate(xs1))
  fill_f32<<<8192, blk, 0, stream>>>(h, 0.0f, (long long)n * HID);
  edge_aggregate<<<ewave, blk, 0, stream>>>(ei, ew, xs, h, E, HID);
  elu_inplace<<<8192, blk, 0, stream>>>(h, (long long)n * HID);

  // h2 = h1 @ conv2_lin
  wmma_gemm_kernel<<<dim3(gx(n), gy(OUT)), blk, 0, stream>>>(
      h, c2lin, nullptr, h2, n, OUT, HID, HID, OUT, OUT, 0, 0);

  // xs3 = h2 @ conv2_lin^T
  wmma_gemm_kernel<<<dim3(gx(n), gy(HID)), blk, 0, stream>>>(
      h2, c2lin, nullptr, xs, n, HID, OUT, OUT, OUT, HID, 1, 0);

  // h3 = elu(aggregate(xs3)) with tied attention weights
  fill_f32<<<8192, blk, 0, stream>>>(h, 0.0f, (long long)n * HID);
  edge_aggregate<<<ewave, blk, 0, stream>>>(ei, ew, xs, h, E, HID);
  elu_inplace<<<8192, blk, 0, stream>>>(h, (long long)n * HID);

  // h4 = h3 @ conv1_lin^T
  wmma_gemm_kernel<<<dim3(gx(n), gy(IN_DIM)), blk, 0, stream>>>(
      h, c1lin, nullptr, h4, n, IN_DIM, HID, HID, HID, IN_DIM, 1, 0);
}
